// SwinTransformerBlock_31473520345482
// MI455X (gfx1250) — matrix operand order repaired and checked by reading, not compiled, not run on hardware
//
#include <hip/hip_runtime.h>
#include <hip/hip_bf16.h>

typedef __attribute__((ext_vector_type(16))) _Float16 v16h;
typedef __attribute__((ext_vector_type(8)))  _Float16 v8h;
typedef __attribute__((ext_vector_type(8)))  float    v8f;
typedef __attribute__((ext_vector_type(4))) unsigned int u32x4;
typedef __attribute__((ext_vector_type(8))) int          i32x8;
typedef __attribute__((ext_vector_type(4))) int          i32x4;

#define DIM   192
#define MLPD  768
#define ODIM  384
#define BATCH 8
#define NTOK  3136          // 56*56
#define MROWS (BATCH*NTOK)  // 25088

// ---------------------------------------------------------------- WMMA helper
__device__ __forceinline__ v8f wmma16(v16h a, v16h b, v8f c) {
  // D = A(16x32 f16) * B(32x16 f16) + C(16x16 f32)
  return __builtin_amdgcn_wmma_f32_16x16x32_f16(false, a, false, b, (short)0, c,
                                                false, false);
}

// A-matrix 16x32 f16 fragment: lane<16 holds K=[0..15], lane>=16 holds
// K=[16..31] of row (lane&15). base already points at this lane's row.
__device__ __forceinline__ v16h load_a_frag(const _Float16* base, int lhalf) {
  v16h r;
  *((v8h*)&r)     = *(const v8h*)(base + lhalf * 16);
  *(((v8h*)&r)+1) = *(const v8h*)(base + lhalf * 16 + 8);
  return r;
}

// -------------------------------------------------- Tensor Data Mover (TDM)
// 2D tile load Global -> LDS via TENSOR_LOAD_TO_LDS (tracked by TENSORcnt).
// D# group0: count=1 | lds_addr[63:32] | global_addr[120:64] | type=2.
// D# group1: data_size=2B, tensor_dim0/1, tile_dim0/1, dim0 stride.
__device__ __forceinline__ void tdm_load_2d(unsigned lds_off,
                                            unsigned long long gaddr,
                                            unsigned dim0, unsigned rows,
                                            unsigned stride) {
  u32x4 g0;
  g0[0] = 1u;                                   // count=1 (valid, user mode)
  g0[1] = lds_off;                              // LDS byte address
  g0[2] = (unsigned)gaddr;                      // global addr [95:64]
  g0[3] = ((unsigned)(gaddr >> 32) & 0x01FFFFFFu) | (2u << 30);  // [120:96]|type=2
  i32x8 g1;
  g1[0] = (int)(1u << 16);                      // workgroup_mask=0, data_size=2B
  g1[1] = (int)((dim0 & 0xFFFFu) << 16);        // tensor_dim0 low16 @ [63:48]
  g1[2] = (int)((dim0 >> 16) | ((rows & 0xFFFFu) << 16));  // dim0 hi | tensor_dim1 lo
  g1[3] = (int)((rows >> 16) | ((dim0 & 0xFFFFu) << 16));  // dim1 hi | tile_dim0
  g1[4] = (int)(rows & 0xFFFFu);                // tile_dim1 (tile_dim2=0)
  g1[5] = (int)stride;                          // tensor_dim0_stride low32
  g1[6] = 0;                                    // stride hi16 | dim1_stride lo16
  g1[7] = 0;
  i32x4 z4 = (i32x4)0;
#if defined(__clang_major__) && (__clang_major__ >= 23)
  i32x8 z8 = (i32x8)0;
  __builtin_amdgcn_tensor_load_to_lds(g0, g1, z4, z4, z8, 0);
#else
  __builtin_amdgcn_tensor_load_to_lds(g0, g1, z4, z4, 0);
#endif
}

__device__ __forceinline__ unsigned lds_offset_of(const void* p) {
  return (unsigned)(unsigned long long)p;  // LDS aperture: addr[31:0] = LDS offset
}

// ------------------------------------------------------- LayerNorm (+reshape)
// from_x=1: read x[B,C,H,W], write t[B,N,C] fp32 and tn fp16
// from_x=0: read tin[B,N,C] fp32, write tn fp16 only
__global__ void ln_kernel(const float* __restrict__ x,
                          const float* __restrict__ tin,
                          float* __restrict__ tout,
                          _Float16* __restrict__ hout,
                          const float* __restrict__ g,
                          const float* __restrict__ be,
                          int from_x) {
  int n = blockIdx.x, b = blockIdx.y, tid = threadIdx.x;  // 64 threads
  float v[3];
#pragma unroll
  for (int j = 0; j < 3; j++) {
    int c = tid + j * 64;
    v[j] = from_x ? x[((size_t)(b * DIM + c)) * NTOK + n]
                  : tin[((size_t)b * NTOK + n) * DIM + c];
  }
  __shared__ float s1[64], s2[64];
  s1[tid] = v[0] + v[1] + v[2];
  s2[tid] = v[0]*v[0] + v[1]*v[1] + v[2]*v[2];
  __syncthreads();
  for (int s = 32; s > 0; s >>= 1) {
    if (tid < s) { s1[tid] += s1[tid + s]; s2[tid] += s2[tid + s]; }
    __syncthreads();
  }
  float mean = s1[0] * (1.0f / DIM);
  float var  = s2[0] * (1.0f / DIM) - mean * mean;
  float inv  = rsqrtf(var + 1e-5f);
  size_t rb = ((size_t)b * NTOK + n) * DIM;
#pragma unroll
  for (int j = 0; j < 3; j++) {
    int c = tid + j * 64;
    float nv = (v[j] - mean) * inv * g[c] + be[c];
    if (from_x) tout[rb + c] = v[j];
    hout[rb + c] = (_Float16)nv;
  }
}

// ----------------------------------------------------------- WMMA GEMM + epi
// out[M,N] = act(A[M,K](f16) * Bp[N,K]^T(f16) + bias) * scale (+ resid[M,DIM])
// block = 128 thr = 4 waves; block tile 64(M) x 64(N); wave tile 16 x 64.
// B tile staged in LDS once per block via TDM; A loads software-pipelined.
__global__ void gemm_kernel(const _Float16* __restrict__ A,
                            const _Float16* __restrict__ Bp,
                            const float* __restrict__ bias,
                            const float* __restrict__ resid,
                            void* __restrict__ out,
                            int K, int N, float scale, int act, int out_f32) {
  extern __shared__ __attribute__((aligned(32))) _Float16 Bs[];  // [64][K]
  int wave = threadIdx.x >> 5, lane = threadIdx.x & 31;
  int l16 = lane & 15, lhalf = lane >> 4;
  int m0 = (blockIdx.x * 4 + wave) * 16;
  int n0 = blockIdx.y * 64;

  if (wave == 0) {
    tdm_load_2d(lds_offset_of(Bs),
                (unsigned long long)(Bp + (size_t)n0 * K),
                (unsigned)K, 64u, (unsigned)K);
    __builtin_amdgcn_s_wait_tensorcnt(0);
  }
  __syncthreads();

  v8f zero8 = {0,0,0,0,0,0,0,0};
  v8f acc[4];
#pragma unroll
  for (int t = 0; t < 4; t++) acc[t] = zero8;

  const _Float16* arow = A + (size_t)(m0 + l16) * K;
  v16h af = load_a_frag(arow, lhalf);                 // prologue k=0
  for (int kc = 32; kc < K; kc += 32) {
    __builtin_prefetch(arow + kc + 32, 0, 1);         // global_prefetch_b8
    v16h afn = load_a_frag(arow + kc, lhalf);         // next k-step (pipelined)
#pragma unroll
    for (int t = 0; t < 4; t++) {
      v16h bf = *(const v16h*)&Bs[(size_t)(t * 16 + l16) * K + (kc - 32) + lhalf * 16];
      acc[t] = wmma16(af, bf, acc[t]);
    }
    af = afn;
  }
#pragma unroll
  for (int t = 0; t < 4; t++) {                       // epilogue k = K-32
    v16h bf = *(const v16h*)&Bs[(size_t)(t * 16 + l16) * K + (K - 32) + lhalf * 16];
    acc[t] = wmma16(af, bf, acc[t]);
  }

#pragma unroll
  for (int t = 0; t < 4; t++) {
#pragma unroll
    for (int i = 0; i < 8; i++) {
      int row = m0 + i + lhalf * 8;
      int col = n0 + t * 16 + l16;
      float val = acc[t][i] + bias[col];
      if (act == 1) val = 0.5f * val * (1.0f + erff(val * 0.70710678f));
      val *= scale;
      if (resid) val += resid[(size_t)row * DIM + col];  // only used when N==DIM
      if (out_f32) ((float*)out)[(size_t)row * N + col] = val;
      else ((_Float16*)out)[(size_t)row * N + col] = (_Float16)val;
    }
  }
}

// ---------------------------------------------------- Flash attention (WMMA)
// grid (49, B), block 128 = 4 waves; wave owns 16 query rows, all 192 out cols.
// Online softmax over 98 chunks of 32 keys. T += softmax(QK^T)V (residual).
// K chunk staged to LDS by TDM (TENSORcnt); V staged transposed by VALU/DS.
__global__ void flash_kernel(const _Float16* __restrict__ Qh,
                             const _Float16* __restrict__ Kh,
                             const _Float16* __restrict__ Vh,
                             float* __restrict__ T) {
  __shared__ _Float16 Ks[32][DIM];    // keys row-major (TDM destination)
  __shared__ _Float16 Vts[DIM][32];   // V transposed: [c][key]
  __shared__ _Float16 Ps[4][16][32];  // per-wave P staging (C/D -> A layout)
  int tid = threadIdx.x;
  int wave = tid >> 5, lane = tid & 31, l16 = lane & 15, lhalf = lane >> 4;
  int b = blockIdx.y;
  int m0 = (blockIdx.x * 4 + wave) * 16;
  size_t bb = (size_t)b * NTOK * DIM;
  unsigned ks_off = lds_offset_of(&Ks[0][0]);

  // resident Q fragments (Q already scaled by 1/sqrt(C) in QKV epilogue)
  v16h qf[6];
  const _Float16* qrow = Qh + bb + (size_t)(m0 + l16) * DIM;
#pragma unroll
  for (int kc = 0; kc < 6; kc++) qf[kc] = load_a_frag(qrow + kc * 32, lhalf);

  v8f zero8 = {0,0,0,0,0,0,0,0};
  v8f acc[12];
#pragma unroll
  for (int t = 0; t < 12; t++) acc[t] = zero8;
  float mrow[8], lrow[8];
#pragma unroll
  for (int i = 0; i < 8; i++) { mrow[i] = -1e30f; lrow[i] = 0.0f; }

  for (int ch = 0; ch < 98; ch++) {
    int key0 = ch * 32;
    // K chunk 32x192 via Tensor Data Mover (issued once, by wave 0)
    if (wave == 0)
      tdm_load_2d(ks_off, (unsigned long long)(Kh + bb + (size_t)key0 * DIM),
                  DIM, 32u, DIM);
    // V chunk staged transposed: 768 v8h units, 6 per thread
    const _Float16* vg = Vh + bb + (size_t)key0 * DIM;
#pragma unroll
    for (int u = 0; u < 6; u++) {
      int unit = tid + u * 128;
      int r = unit / 24, off = (unit % 24) * 8;
      v8h vv = *(const v8h*)(vg + (size_t)r * DIM + off);
#pragma unroll
      for (int j = 0; j < 8; j++) Vts[off + j][r] = vv[j];
    }
    if (wave == 0) __builtin_amdgcn_s_wait_tensorcnt(0);
    __syncthreads();

    // S(16x32) = Q(16x192) * K_chunk^T
    v8f s0 = zero8, s1 = zero8;
#pragma unroll
    for (int kc = 0; kc < 6; kc++) {
      v16h k0 = *(const v16h*)&Ks[l16][kc * 32 + lhalf * 16];
      v16h k1 = *(const v16h*)&Ks[16 + l16][kc * 32 + lhalf * 16];
      s0 = wmma16(qf[kc], k0, s0);
      s1 = wmma16(qf[kc], k1, s1);
    }

    // online softmax; rows map to (vgpr i, lane half); reduce within 16 lanes
    float alpha[8];
#pragma unroll
    for (int i = 0; i < 8; i++) {
      float a0 = s0[i], a1 = s1[i];
      float mx = fmaxf(a0, a1);
#pragma unroll
      for (int d = 1; d < 16; d <<= 1) mx = fmaxf(mx, __shfl_xor(mx, d, 32));
      float mnew = fmaxf(mrow[i], mx);
      alpha[i] = __expf(mrow[i] - mnew);
      float p0 = __expf(a0 - mnew), p1 = __expf(a1 - mnew);
      float rs = p0 + p1;
#pragma unroll
      for (int d = 1; d < 16; d <<= 1) rs += __shfl_xor(rs, d, 32);
      lrow[i] = lrow[i] * alpha[i] + rs;
      mrow[i] = mnew;
      int prow = i + lhalf * 8;
      Ps[wave][prow][l16]      = (_Float16)p0;
      Ps[wave][prow][16 + l16] = (_Float16)p1;
    }
#pragma unroll
    for (int t = 0; t < 12; t++)
#pragma unroll
      for (int i = 0; i < 8; i++) acc[t][i] = acc[t][i] * alpha[i];

    // P (C/D layout) -> A fragment via same-wave LDS round trip (in-order DS)
    v16h pf = load_a_frag(&Ps[wave][l16][0], lhalf);
#pragma unroll
    for (int t = 0; t < 12; t++) {
      v16h vf = *(const v16h*)&Vts[t * 16 + l16][lhalf * 16];
      acc[t] = wmma16(pf, vf, acc[t]);
    }
    __syncthreads();  // protect Ks/Vts before next chunk's TDM/stores
  }

  // epilogue: O = acc/l, residual add into t
#pragma unroll
  for (int i = 0; i < 8; i++) {
    float invl = 1.0f / lrow[i];
    int row = m0 + i + lhalf * 8;
    float* trow = T + bb + (size_t)row * DIM;
#pragma unroll
    for (int t = 0; t < 12; t++) trow[t * 16 + l16] += acc[t][i] * invl;
  }
}

// --------------------------------------------------------------- aux kernels
__global__ void cvt_h_kernel(const float* __restrict__ s,
                             _Float16* __restrict__ d, int n) {
  int i = blockIdx.x * blockDim.x + threadIdx.x;
  if (i < n) d[i] = (_Float16)s[i];
}

__global__ void pool_kernel(const float* __restrict__ y, float* __restrict__ out) {
  int idx = blockIdx.x * blockDim.x + threadIdx.x;
  if (idx >= BATCH * ODIM * 49) return;
  int ow = idx % 7, oh = (idx / 7) % 7, o = (idx / 49) % ODIM, b = idx / (49 * ODIM);
  float s = 0.0f;
  for (int r = 0; r < 8; r++)
    for (int c = 0; c < 8; c++) {
      int n = (oh * 8 + r) * 56 + ow * 8 + c;
      s += y[((size_t)b * NTOK + n) * ODIM + o];
    }
  out[idx] = s * (1.0f / 64.0f);
}

// ------------------------------------------------------------------- launch
extern "C" void kernel_launch(void* const* d_in, const int* in_sizes, int n_in,
                              void* d_out, int out_size, void* d_ws, size_t ws_size,
                              hipStream_t stream) {
  (void)in_sizes; (void)n_in; (void)out_size; (void)ws_size;
  const float* x    = (const float*)d_in[0];
  const float* Wq   = (const float*)d_in[1];  const float* bq   = (const float*)d_in[2];
  const float* Wk   = (const float*)d_in[3];  const float* bk   = (const float*)d_in[4];
  const float* Wv   = (const float*)d_in[5];  const float* bv   = (const float*)d_in[6];
  const float* g1   = (const float*)d_in[7];  const float* b1   = (const float*)d_in[8];
  const float* g2   = (const float*)d_in[9];  const float* b2   = (const float*)d_in[10];
  const float* Wfc1 = (const float*)d_in[11]; const float* bfc1 = (const float*)d_in[12];
  const float* Wfc2 = (const float*)d_in[13]; const float* bfc2 = (const float*)d_in[14];
  const float* Wcv  = (const float*)d_in[15]; const float* bcv  = (const float*)d_in[16];

  char* ws = (char*)d_ws;
  float*    t    = (float*)   (ws + 0);         // 19,267,584 B
  _Float16* tn   = (_Float16*)(ws + 19267584);  //  9,633,792 B (reused for LN2)
  _Float16* Qh   = (_Float16*)(ws + 28901376);
  _Float16* Kh   = (_Float16*)(ws + 38535168);
  _Float16* Vh   = (_Float16*)(ws + 48168960);
  _Float16* Th   = (_Float16*)(ws + 48168960);  // alias V (dead after attention)
  _Float16* Wq_h = (_Float16*)(ws + 57802752);
  _Float16* Wk_h = (_Float16*)(ws + 57876480);
  _Float16* Wv_h = (_Float16*)(ws + 57950208);
  _Float16* Wf1h = (_Float16*)(ws + 58023936);
  _Float16* Wf2h = (_Float16*)(ws + 58318848);
  _Float16* Wcvh = (_Float16*)(ws + 58613760);
  _Float16* hdd  = (_Float16*)(ws + 58761216);  // 38,535,168 B
  float*    y    = (float*)   (ws + 58761216);  // alias hdd (dead after MLP2)

  // fp16 weight packs (row-major == N-major/K-contig B operand)
  cvt_h_kernel<<<(36864  + 255) / 256, 256, 0, stream>>>(Wq,   Wq_h, 36864);
  cvt_h_kernel<<<(36864  + 255) / 256, 256, 0, stream>>>(Wk,   Wk_h, 36864);
  cvt_h_kernel<<<(36864  + 255) / 256, 256, 0, stream>>>(Wv,   Wv_h, 36864);
  cvt_h_kernel<<<(147456 + 255) / 256, 256, 0, stream>>>(Wfc1, Wf1h, 147456);
  cvt_h_kernel<<<(147456 + 255) / 256, 256, 0, stream>>>(Wfc2, Wf2h, 147456);
  cvt_h_kernel<<<(73728  + 255) / 256, 256, 0, stream>>>(Wcv,  Wcvh, 73728);

  dim3 lng(NTOK, BATCH);
  ln_kernel<<<lng, 64, 0, stream>>>(x, nullptr, t, tn, g1, b1, 1);

  const float qscale = 0.072168784f;  // 1/sqrt(192)
  dim3 gq(MROWS / 64, DIM / 64);      // 392 x 3
  size_t bs192 = (size_t)64 * 192 * 2, bs768 = (size_t)64 * 768 * 2;
  gemm_kernel<<<gq, 128, bs192, stream>>>(tn, Wq_h, bq, nullptr, Qh, DIM, DIM, qscale, 0, 0);
  gemm_kernel<<<gq, 128, bs192, stream>>>(tn, Wk_h, bk, nullptr, Kh, DIM, DIM, 1.0f, 0, 0);
  gemm_kernel<<<gq, 128, bs192, stream>>>(tn, Wv_h, bv, nullptr, Vh, DIM, DIM, 1.0f, 0, 0);

  dim3 gf(NTOK / 64, BATCH);          // 49 x 8
  flash_kernel<<<gf, 128, 0, stream>>>(Qh, Kh, Vh, t);

  ln_kernel<<<lng, 64, 0, stream>>>(nullptr, t, nullptr, tn, g2, b2, 0);

  dim3 gm1(MROWS / 64, MLPD / 64);    // 392 x 12
  gemm_kernel<<<gm1, 128, bs192, stream>>>(tn, Wf1h, bfc1, nullptr, hdd, DIM, MLPD, 1.0f, 1, 0);
  dim3 gm2(MROWS / 64, DIM / 64);     // 392 x 3
  gemm_kernel<<<gm2, 128, bs768, stream>>>(hdd, Wf2h, bfc2, t, Th, MLPD, DIM, 1.0f, 0, 0);
  dim3 gcv(MROWS / 64, ODIM / 64);    // 392 x 6
  gemm_kernel<<<gcv, 128, bs192, stream>>>(Th, Wcvh, bcv, nullptr, y, DIM, ODIM, 1.0f, 0, 1);

  pool_kernel<<<(BATCH * ODIM * 49 + 255) / 256, 256, 0, stream>>>(y, (float*)d_out);
}
